// DKM_param_3367254360578
// MI455X (gfx1250) — matrix operand order repaired and checked by reading, compiled, not run on hardware
//
#include <hip/hip_runtime.h>
#include <hip/hip_bf16.h>
#include <math.h>

// ---------------------------------------------------------------------------
// Problem constants (from reference)
// ---------------------------------------------------------------------------
#define N_PTS 2048
#define M_CL  32
#define D_DIM 768
#define H_DIM 512
#define K2D   (2 * D_DIM)          // 1536
#define INV_TEMP 20.0f             // 1/0.05
#define THRESH 1e-4f
#define EPS_F  1e-6f

typedef __attribute__((ext_vector_type(16))) __bf16 v16bf;
typedef __attribute__((ext_vector_type(8)))  float  v8f;

// ---------------------------------------------------------------------------
// Workspace layout (bytes, all 256-aligned). Total ~12.7 MB.
// ---------------------------------------------------------------------------
#define OFF_U      0u                                  // N x 2D bf16
#define OFF_WCATT  (OFF_U     + N_PTS * K2D * 2u)      // H x 2D bf16
#define OFF_W1XYT  (OFF_WCATT + H_DIM * K2D * 2u)      // H x D bf16
#define OFF_APRE   (OFF_W1XYT + H_DIM * D_DIM * 2u)    // N x H f32
#define OFF_BPRE   (OFF_APRE  + (size_t)N_PTS * H_DIM * 4u) // M x H f32
#define OFF_DIST   (OFF_BPRE  + M_CL * H_DIM * 4u)     // N x M f32
#define OFF_CNEW   (OFF_DIST  + (size_t)N_PTS * M_CL * 4u)  // M x D f32
#define OFF_ASUM   (OFF_CNEW  + M_CL * D_DIM * 4u)     // M f32
#define OFF_DIFF   (OFF_ASUM  + 256u)                  // 1 f32

// ---------------------------------------------------------------------------
// WMMA fragment loaders (CDNA5 ISA VGPR layouts, wave32).
// A (16x32 bf16, row M striped in-lane): lanes 0-15 carry K {0-7,16-23},
// lanes 16-31 carry K {8-15,24-31}, row = lane&15.
// B (32x16 bf16, from TRANSPOSED weights Wt[H][K]): col = lane&15,
// lanes 0-15 hold K {0-7,16-23}, lanes 16-31 hold K {8-15,24-31} (same as A).
// ---------------------------------------------------------------------------
__device__ __forceinline__ v16bf load_frag_a(const __bf16* row_base, int k0) {
    int lane = threadIdx.x & 31;
    int off  = (lane >> 4) << 3;                 // 0 or 8
    const __bf16* p = row_base + k0 + off;       // 16B aligned
    union { uint4 u[2]; v16bf v; } f;
    f.u[0] = *(const uint4*)(p);                 // K chunk 0
    f.u[1] = *(const uint4*)(p + 16);            // K chunk +16
    return f.v;
}

__device__ __forceinline__ v16bf load_frag_bT(const __bf16* Wt, int ldk, int n0, int k0) {
    int lane = threadIdx.x & 31;
    int col  = n0 + (lane & 15);
    int kk   = k0 + ((lane >> 4) << 3);          // 0 or 8
    const __bf16* p = Wt + (size_t)col * ldk + kk; // two 16B chunks
    union { uint4 u[2]; v16bf v; } f;
    f.u[0] = *(const uint4*)(p);
    f.u[1] = *(const uint4*)(p + 16);
    return f.v;
}

// ---------------------------------------------------------------------------
// Pack kernels: build bf16 operands once (reused by every pairwise tile).
// ---------------------------------------------------------------------------
__global__ void pack_U_kernel(const float* __restrict__ X,
                              const float* __restrict__ q,
                              __bf16* __restrict__ U) {
    int idx = blockIdx.x * blockDim.x + threadIdx.x;       // over N*D
    if (idx >= N_PTS * D_DIM) return;
    int i = idx / D_DIM, d = idx % D_DIM;
    float x = X[idx];
    U[(size_t)i * K2D + d]         = (__bf16)x;               // x block
    U[(size_t)i * K2D + D_DIM + d] = (__bf16)fabsf(x - q[d]); // |x-q| block
}

__global__ void pack_W_kernel(const float* __restrict__ W1,
                              __bf16* __restrict__ WcatT,
                              __bf16* __restrict__ W1xyT) {
    int idx = blockIdx.x * blockDim.x + threadIdx.x;       // over H*D
    if (idx >= H_DIM * D_DIM) return;
    int h = idx / D_DIM, d = idx % D_DIM;
    // W1 is (5D, H) row-major: blocks [x, y, xy, xq, yq]
    WcatT[(size_t)h * K2D + d]         = (__bf16)W1[(size_t)d * H_DIM + h];
    WcatT[(size_t)h * K2D + D_DIM + d] = (__bf16)W1[(size_t)(3 * D_DIM + d) * H_DIM + h];
    W1xyT[(size_t)h * D_DIM + d]       = (__bf16)W1[(size_t)(2 * D_DIM + d) * H_DIM + h];
}

// Per-j bias row: Bpre(j,h) = b1[h] + y_j.W1y + |y_j - q|.W1yq  (tiny: 32x512)
__global__ void bpre_kernel(const float* __restrict__ Cinit,
                            const float* __restrict__ q,
                            const float* __restrict__ W1,
                            const float* __restrict__ b1,
                            float* __restrict__ Bpre) {
    int j = blockIdx.x, h = threadIdx.x;
    float s = b1[h];
    for (int d = 0; d < D_DIM; ++d) {
        float y = Cinit[(size_t)j * D_DIM + d];
        s += y * W1[(size_t)(D_DIM + d) * H_DIM + h];
        s += fabsf(y - q[d]) * W1[(size_t)(4 * D_DIM + d) * H_DIM + h];
    }
    Bpre[(size_t)j * H_DIM + h] = s;
}

// ---------------------------------------------------------------------------
// Apre = U @ WcatT^T  (2048x1536 @ 1536x512, bf16 WMMA, f32 accum)
// block = 4 waves; wave w -> two adjacent 16-wide h-tiles (A-frag reused by
// two WMMAs -> 3 operand loads per WMMA instead of 4). Block covers 128 h.
// ---------------------------------------------------------------------------
__global__ __launch_bounds__(128) void gemm_apre_kernel(
    const __bf16* __restrict__ U, const __bf16* __restrict__ WcatT,
    float* __restrict__ Apre) {
    int wave = threadIdx.x >> 5;
    int lane = threadIdx.x & 31;
    int i0 = blockIdx.x * 16;
    int h0 = blockIdx.y * 128 + wave * 32;       // tiles h0 and h0+16
    const __bf16* arow = U + (size_t)(i0 + (lane & 15)) * K2D;
    v8f c0 = {}, c1 = {};
    for (int k0 = 0; k0 < K2D; k0 += 32) {
        v16bf a  = load_frag_a(arow, k0);
        v16bf b0 = load_frag_bT(WcatT, K2D, h0,      k0);
        v16bf b1v = load_frag_bT(WcatT, K2D, h0 + 16, k0);
        c0 = __builtin_amdgcn_wmma_f32_16x16x32_bf16(false, a, false, b0,
                                                     (short)0, c0, false, false);
        c1 = __builtin_amdgcn_wmma_f32_16x16x32_bf16(false, a, false, b1v,
                                                     (short)0, c1, false, false);
    }
    int n = lane & 15;
    int mbase = (lane >> 4) << 3;
#pragma unroll
    for (int r = 0; r < 8; ++r) {
        Apre[(size_t)(i0 + mbase + r) * H_DIM + h0 + n]      = c0[r];
        Apre[(size_t)(i0 + mbase + r) * H_DIM + h0 + 16 + n] = c1[r];
    }
}

// ---------------------------------------------------------------------------
// Pairwise kernel: grid (N/16, M). Per block: 16 i-rows x 1 j.
//   LDS tile |x_i - y_j| (16 x 768 bf16, 24KB) built once, 4 waves sweep the
//   32 h-tiles in pairs (A-frag from LDS reused by two WMMAs), fused
//   +Apre +Bpre, ReLU, dot W2, fixed-order reduce, sigmoid -> dist.
// ---------------------------------------------------------------------------
#define PW_WAVES 4
__global__ __launch_bounds__(128) void pairwise_kernel(
    const float* __restrict__ X, const float* __restrict__ Cinit,
    const __bf16* __restrict__ W1xyT,
    const float* __restrict__ Apre, const float* __restrict__ Bpre,
    const float* __restrict__ W2, const float* __restrict__ b2,
    float* __restrict__ dist) {
    __shared__ __align__(16) __bf16 ldsAbs[16 * D_DIM];
    __shared__ float sred[PW_WAVES * 32 * 8];

    int i0 = blockIdx.x * 16;
    int j  = blockIdx.y;
    int tid = threadIdx.x;

    // Build the pairwise A-operand tile in LDS (never touches HBM).
    for (int e = tid; e < 16 * D_DIM; e += blockDim.x) {
        int r = e / D_DIM, d = e % D_DIM;
        float v = fabsf(X[(size_t)(i0 + r) * D_DIM + d] - Cinit[(size_t)j * D_DIM + d]);
        ldsAbs[e] = (__bf16)v;
    }
    __syncthreads();

    int wave = tid >> 5, lane = tid & 31;
    const __bf16* arow = ldsAbs + (size_t)(lane & 15) * D_DIM;
    int n = lane & 15;
    int mbase = (lane >> 4) << 3;

    float sacc[8];
#pragma unroll
    for (int r = 0; r < 8; ++r) sacc[r] = 0.f;

    // 32 h-tiles -> 16 pairs; 4 waves x 4 pairs each.
    for (int p = wave; p < H_DIM / 32; p += PW_WAVES) {
        int h0 = p * 32;                          // tiles h0 and h0+16
        v8f c0 = {}, c1 = {};
        for (int k0 = 0; k0 < D_DIM; k0 += 32) {
            v16bf a  = load_frag_a(arow, k0);
            v16bf b0 = load_frag_bT(W1xyT, D_DIM, h0,      k0);
            v16bf b1v = load_frag_bT(W1xyT, D_DIM, h0 + 16, k0);
            c0 = __builtin_amdgcn_wmma_f32_16x16x32_bf16(false, a, false, b0,
                                                         (short)0, c0, false, false);
            c1 = __builtin_amdgcn_wmma_f32_16x16x32_bf16(false, a, false, b1v,
                                                         (short)0, c1, false, false);
        }
        int ha = h0 + n, hb = h0 + 16 + n;
        float w2a = W2[ha],                 w2b = W2[hb];
        float bja = Bpre[(size_t)j * H_DIM + ha];
        float bjb = Bpre[(size_t)j * H_DIM + hb];
#pragma unroll
        for (int r = 0; r < 8; ++r) {
            size_t irow = (size_t)(i0 + mbase + r) * H_DIM;
            float pa = c0[r] + Apre[irow + ha] + bja;
            float pb = c1[r] + Apre[irow + hb] + bjb;
            sacc[r] += fmaxf(pa, 0.f) * w2a + fmaxf(pb, 0.f) * w2b;
        }
    }
#pragma unroll
    for (int r = 0; r < 8; ++r) sred[(wave * 32 + lane) * 8 + r] = sacc[r];
    __syncthreads();

    // Deterministic fixed-order reduction over (waves x 16 lanes) per row.
    if (tid < 16) {
        int m = tid;
        int rr = m & 7;
        int lbase = (m >> 3) << 4;   // lanes 0..15 hold rows 0-7; 16..31 rows 8-15
        float s = 0.f;
        for (int w = 0; w < PW_WAVES; ++w)
            for (int l = 0; l < 16; ++l)
                s += sred[(w * 32 + lbase + l) * 8 + rr];
        s += b2[0];
        float sig = 1.f / (1.f + expf(-s));     // learned distance in (0,1)
        dist[(size_t)(i0 + m) * M_CL + j] = sig;
    }
}

// ---------------------------------------------------------------------------
// Softmax over j (M=32) per row: a = softmax(-dist/TEMP). One thread per row.
// ---------------------------------------------------------------------------
__global__ void softmax_kernel(const float* __restrict__ dist,
                               float* __restrict__ aout) {
    int i = blockIdx.x * blockDim.x + threadIdx.x;
    if (i >= N_PTS) return;
    float v[M_CL];
    float mx = -1e30f;
#pragma unroll
    for (int j = 0; j < M_CL; ++j) {
        v[j] = -dist[(size_t)i * M_CL + j] * INV_TEMP;
        mx = fmaxf(mx, v[j]);
    }
    float s = 0.f;
#pragma unroll
    for (int j = 0; j < M_CL; ++j) { v[j] = expf(v[j] - mx); s += v[j]; }
    float inv = 1.f / s;
#pragma unroll
    for (int j = 0; j < M_CL; ++j) aout[(size_t)i * M_CL + j] = v[j] * inv;
}

// a_sum (M) — single block, fixed-order two-level reduction.
__global__ void asum_kernel(const float* __restrict__ a, float* __restrict__ asum) {
    __shared__ float part[256];
    int t = threadIdx.x;
    int col = t & 31, chunk = t >> 5;            // 8 chunks of rows
    float s = 0.f;
    for (int i = chunk; i < N_PTS; i += 8) s += a[(size_t)i * M_CL + col];
    part[t] = s;
    __syncthreads();
    if (t < 32) {
        float tot = 0.f;
        for (int c = 0; c < 8; ++c) tot += part[c * 32 + t];
        asum[t] = tot + EPS_F;
    }
}

// C_new = (a^T @ X) / a_sum  (32x768, K=2048)
__global__ void cnew_kernel(const float* __restrict__ a, const float* __restrict__ X,
                            const float* __restrict__ asum, float* __restrict__ Cnew) {
    int idx = blockIdx.x * blockDim.x + threadIdx.x;
    if (idx >= M_CL * D_DIM) return;
    int jm = idx / D_DIM, d = idx % D_DIM;
    float s = 0.f;
    for (int i = 0; i < N_PTS; ++i)
        s += a[(size_t)i * M_CL + jm] * X[(size_t)i * D_DIM + d];
    Cnew[idx] = s / asum[jm];
}

// diff = sum |C_new - C_init| — single-block deterministic tree reduction.
__global__ void diff_kernel(const float* __restrict__ Cnew,
                            const float* __restrict__ Cinit,
                            float* __restrict__ diff) {
    __shared__ float part[1024];
    int t = threadIdx.x;
    float s = 0.f;
    for (int e = t; e < M_CL * D_DIM; e += 1024) s += fabsf(Cnew[e] - Cinit[e]);
    part[t] = s;
    __syncthreads();
    for (int off = 512; off > 0; off >>= 1) {
        if (t < off) part[t] += part[t + off];
        __syncthreads();
    }
    if (t == 0) *diff = part[0];
}

__global__ void select_kernel(const float* __restrict__ Cnew,
                              const float* __restrict__ Cinit,
                              const float* __restrict__ diff,
                              float* __restrict__ out) {
    int idx = blockIdx.x * blockDim.x + threadIdx.x;
    if (idx >= M_CL * D_DIM) return;
    out[idx] = (*diff > THRESH) ? Cnew[idx] : Cinit[idx];
}

// ---------------------------------------------------------------------------
// Launcher
// ---------------------------------------------------------------------------
extern "C" void kernel_launch(void* const* d_in, const int* in_sizes, int n_in,
                              void* d_out, int out_size, void* d_ws, size_t ws_size,
                              hipStream_t stream) {
    (void)in_sizes; (void)n_in; (void)out_size; (void)ws_size;
    const float* q     = (const float*)d_in[0];
    const float* X     = (const float*)d_in[1];
    const float* Cinit = (const float*)d_in[2];
    const float* W1    = (const float*)d_in[3];
    const float* b1    = (const float*)d_in[4];
    const float* W2    = (const float*)d_in[5];
    const float* b2    = (const float*)d_in[6];

    char* ws = (char*)d_ws;
    __bf16* U      = (__bf16*)(ws + OFF_U);
    __bf16* WcatT  = (__bf16*)(ws + OFF_WCATT);
    __bf16* W1xyT  = (__bf16*)(ws + OFF_W1XYT);
    float*  Apre   = (float*)(ws + OFF_APRE);
    float*  Bpre   = (float*)(ws + OFF_BPRE);
    float*  dist   = (float*)(ws + OFF_DIST);
    float*  Cnew   = (float*)(ws + OFF_CNEW);
    float*  asum   = (float*)(ws + OFF_ASUM);
    float*  diff   = (float*)(ws + OFF_DIFF);

    float* Cfinal = (float*)d_out;                 // M*D floats
    float* a_out  = (float*)d_out + M_CL * D_DIM;  // N*M floats

    // 1) pack bf16 operands
    pack_U_kernel<<<(N_PTS * D_DIM + 255) / 256, 256, 0, stream>>>(X, q, U);
    pack_W_kernel<<<(H_DIM * D_DIM + 255) / 256, 256, 0, stream>>>(W1, WcatT, W1xyT);
    // 2) per-j bias rows
    bpre_kernel<<<M_CL, H_DIM, 0, stream>>>(Cinit, q, W1, b1, Bpre);
    // 3) per-i term: Apre = U @ WcatT^T via WMMA (2 h-tiles per wave)
    gemm_apre_kernel<<<dim3(N_PTS / 16, H_DIM / 128), 128, 0, stream>>>(U, WcatT, Apre);
    // 4) pairwise cross-term + MLP head via WMMA (2 h-tiles per wave)
    pairwise_kernel<<<dim3(N_PTS / 16, M_CL), 128, 0, stream>>>(
        X, Cinit, W1xyT, Apre, Bpre, W2, b2, dist);
    // 5) softmax -> a
    softmax_kernel<<<(N_PTS + 255) / 256, 256, 0, stream>>>(dist, a_out);
    // 6) column sums
    asum_kernel<<<1, 256, 0, stream>>>(a_out, asum);
    // 7) C_new
    cnew_kernel<<<(M_CL * D_DIM + 255) / 256, 256, 0, stream>>>(a_out, X, asum, Cnew);
    // 8) L1 diff
    diff_kernel<<<1, 1024, 0, stream>>>(Cnew, Cinit, diff);
    // 9) threshold select
    select_kernel<<<(M_CL * D_DIM + 255) / 256, 256, 0, stream>>>(Cnew, Cinit, diff, Cfinal);
}
